// SpaceFormerCelltype_1400159338534
// MI455X (gfx1250) — matrix operand order repaired and checked by reading, not compiled, not run on hardware
//
#include <hip/hip_runtime.h>
#include <cstdint>

// ---------------------------------------------------------------------------
// Types for WMMA bf16 path (gfx1250)
// ---------------------------------------------------------------------------
typedef __attribute__((ext_vector_type(16))) __bf16 v16bf;
typedef __attribute__((ext_vector_type(8)))  float  v8f;

union BfFrag {
    uint4 u4[2];
    v16bf v;
};

__device__ __forceinline__ uint16_t f2bf(float f) {
    uint32_t u = __float_as_uint(f);
    uint32_t r = u + 0x7FFFu + ((u >> 16) & 1u);
    return (uint16_t)(r >> 16);
}

// ---------------------------------------------------------------------------
// Constants
// ---------------------------------------------------------------------------
#define SF_N    4096
#define SF_D    512
#define SF_FFN  2048
#define SF_HID  256
#define SF_OUT  64
#define SF_GAMMA 0.5f
#define SF_EPS   1e-5f
#define SF_SCALE 0.04419417382415922f   // 1/sqrt(512)

#define LDT 40   // padded LDS row stride (bf16 elems) -> 80B, conflict-free

enum { EPI_F32 = 0, EPI_BF16 = 1, EPI_MASK = 2, EPI_TBF16 = 3 };

// async global->LDS copy of 16 bytes per lane (CDNA5; ASYNCcnt-tracked)
__device__ __forceinline__ void async_cp_b128(uint32_t lds_off, uint64_t gaddr) {
    asm volatile("global_load_async_to_lds_b128 %0, %1, off"
                 :
                 : "v"(lds_off), "v"(gaddr)
                 : "memory");
}

// ---------------------------------------------------------------------------
// NT GEMM: C[M,N] = A[M,K](bf16,row-major) * Bt[N,K](bf16,row-major)^T
// 256 threads = 8 waves; 128x128 tile; each wave owns 32x64 (2x4 WMMA tiles).
// Double-buffered LDS staged via global_load_async_to_lds_b128.
// ---------------------------------------------------------------------------
template <int EPI, int ACT>
__global__ __launch_bounds__(256) void gemm_nt(
    const uint16_t* __restrict__ A, const uint16_t* __restrict__ Bt,
    void* __restrict__ Cout, const float* __restrict__ bias,
    const unsigned char* __restrict__ mask, float scale,
    int M, int Ncols, int K, int ldOutT)
{
    __shared__ uint16_t As[2][128 * LDT];
    __shared__ uint16_t Bs[2][128 * LDT];

    const int tid  = threadIdx.x;
    const int lane = tid & 31;
    const int w    = tid >> 5;
    const int wr   = w & 3;     // 4 wave-rows  * 32 rows
    const int wc   = w >> 2;    // 2 wave-cols  * 64 cols
    const int l16  = lane & 15;
    const int lhi  = lane >> 4;

    const int bm = blockIdx.y * 128;
    const int bn = blockIdx.x * 128;

    v8f acc[2][4];
#pragma unroll
    for (int mi = 0; mi < 2; ++mi)
#pragma unroll
        for (int ni = 0; ni < 4; ++ni)
            acc[mi][ni] = (v8f){0.f, 0.f, 0.f, 0.f, 0.f, 0.f, 0.f, 0.f};

    // stage one 128x32 bf16 tile pair into LDS buffer `buf` (4 async ops/wave)
    auto stage = [&](int k0, int buf) {
#pragma unroll
        for (int it = 0; it < 2; ++it) {
            const int ch = tid + it * 256;
            const int r  = ch >> 2;
            const int c  = (ch & 3) * 8;
            const uint64_t ga =
                (uint64_t)(uintptr_t)(A + (size_t)(bm + r) * K + k0 + c);
            async_cp_b128((uint32_t)(uintptr_t)&As[buf][r * LDT + c], ga);
            // clamp OOB rows (their outputs are never stored)
            const int rB = (bn + r < Ncols) ? (bn + r) : (Ncols - 1);
            const uint64_t gb =
                (uint64_t)(uintptr_t)(Bt + (size_t)rB * K + k0 + c);
            async_cp_b128((uint32_t)(uintptr_t)&Bs[buf][r * LDT + c], gb);
        }
    };

    const int nk = K >> 5;
    stage(0, 0);

    for (int ki = 0; ki < nk; ++ki) {
        const int buf = ki & 1;
        const bool hasNext = (ki + 1) < nk;
        if (hasNext) {
            stage((ki + 1) << 5, buf ^ 1);
            // 4 ops just issued remain outstanding; everything older is done
            asm volatile("s_wait_asynccnt 0x4" ::: "memory");
            if ((ki + 2) < nk)
                __builtin_prefetch(A + (size_t)(bm + (tid >> 1)) * K + ((ki + 2) << 5), 0, 3);
        } else {
            asm volatile("s_wait_asynccnt 0x0" ::: "memory");
        }
        __syncthreads();   // tile `buf` fully resident for all waves

        BfFrag a[2], b[4];
#pragma unroll
        for (int mi = 0; mi < 2; ++mi) {
            const uint16_t* p = &As[buf][(wr * 32 + mi * 16 + l16) * LDT];
            a[mi].u4[0] = *(const uint4*)(p + lhi * 8);
            a[mi].u4[1] = *(const uint4*)(p + 16 + lhi * 8);
        }
#pragma unroll
        for (int ni = 0; ni < 4; ++ni) {
            const uint16_t* p = &Bs[buf][(wc * 64 + ni * 16 + l16) * LDT];
            b[ni].u4[0] = *(const uint4*)(p + lhi * 8);
            b[ni].u4[1] = *(const uint4*)(p + 16 + lhi * 8);
        }
        __syncthreads();   // frags in regs; buf may be overwritten next iter

#pragma unroll
        for (int mi = 0; mi < 2; ++mi)
#pragma unroll
            for (int ni = 0; ni < 4; ++ni)
                acc[mi][ni] = __builtin_amdgcn_wmma_f32_16x16x32_bf16(
                    false, a[mi].v, false, b[ni].v, (short)0, acc[mi][ni],
                    false, false);
    }

    // epilogue
#pragma unroll
    for (int mi = 0; mi < 2; ++mi) {
#pragma unroll
        for (int ni = 0; ni < 4; ++ni) {
            const int col = bn + wc * 64 + ni * 16 + l16;
            if (col >= Ncols) continue;
            float bv = 0.f;
            if (EPI == EPI_F32 || EPI == EPI_BF16)
                if (bias) bv = bias[col];
#pragma unroll
            for (int v = 0; v < 8; ++v) {
                const int row = bm + wr * 32 + mi * 16 + v + 8 * lhi;
                float val = acc[mi][ni][v];
                if constexpr (EPI == EPI_MASK) {
                    val *= scale;
                    if (mask[(size_t)row * Ncols + col]) val = -1e9f;
                    ((float*)Cout)[(size_t)row * Ncols + col] = val;
                } else if constexpr (EPI == EPI_TBF16) {
                    ((uint16_t*)Cout)[(size_t)col * ldOutT + row] = f2bf(val);
                } else {
                    val += bv;
                    if (ACT == 1)
                        val = 0.5f * val * (1.0f + erff(val * 0.7071067811865475f));
                    else if (ACT == 2)
                        val = fmaxf(val, 0.f);
                    if constexpr (EPI == EPI_F32)
                        ((float*)Cout)[(size_t)row * Ncols + col] = val;
                    else
                        ((uint16_t*)Cout)[(size_t)row * Ncols + col] = f2bf(val);
                }
            }
        }
    }
}

// ---------------------------------------------------------------------------
// Dual softmax: p = exp(sr-m) + gamma*exp(sf-m), m = shared row max; normalize
// one block per row; N = 4096, 256 threads -> 16 cols/thread per matrix
// ---------------------------------------------------------------------------
__global__ __launch_bounds__(256) void softmax_dual(
    const float* __restrict__ Sr, const float* __restrict__ Sf,
    uint16_t* __restrict__ P)
{
    const int row = blockIdx.x;
    const int tid = threadIdx.x;
    const float* sr = Sr + (size_t)row * SF_N;
    const float* sf = Sf + (size_t)row * SF_N;
    uint16_t*    pr = P + (size_t)row * SF_N;

    __shared__ float red[256];

    float vr[16], vf[16];
    float mx = -3.4e38f;
#pragma unroll
    for (int i = 0; i < 16; ++i) {
        const int c = tid + i * 256;
        vr[i] = sr[c];
        vf[i] = sf[c];
        mx = fmaxf(mx, fmaxf(vr[i], vf[i]));
    }
    red[tid] = mx;
    __syncthreads();
    for (int s = 128; s > 0; s >>= 1) {
        if (tid < s) red[tid] = fmaxf(red[tid], red[tid + s]);
        __syncthreads();
    }
    mx = red[0];
    __syncthreads();

    float sum = 0.f;
#pragma unroll
    for (int i = 0; i < 16; ++i) {
        const float p = __expf(vr[i] - mx) + SF_GAMMA * __expf(vf[i] - mx);
        vr[i] = p;
        sum += p;
    }
    red[tid] = sum;
    __syncthreads();
    for (int s = 128; s > 0; s >>= 1) {
        if (tid < s) red[tid] += red[tid + s];
        __syncthreads();
    }
    const float inv = 1.0f / red[0];
#pragma unroll
    for (int i = 0; i < 16; ++i)
        pr[tid + i * 256] = f2bf(vr[i] * inv);
}

// ---------------------------------------------------------------------------
// LayerNorm(a + b) with gamma/beta; writes f32 and bf16 copies. D = 512.
// ---------------------------------------------------------------------------
__global__ __launch_bounds__(256) void ln_residual(
    const float* __restrict__ a, const float* __restrict__ b,
    const float* __restrict__ g, const float* __restrict__ be,
    float* __restrict__ outf, uint16_t* __restrict__ outb)
{
    const int row = blockIdx.x;
    const int tid = threadIdx.x;
    const size_t base = (size_t)row * SF_D;

    __shared__ float red[256];

    const float x0 = a[base + tid] + b[base + tid];
    const float x1 = a[base + tid + 256] + b[base + tid + 256];

    red[tid] = x0 + x1;
    __syncthreads();
    for (int s = 128; s > 0; s >>= 1) {
        if (tid < s) red[tid] += red[tid + s];
        __syncthreads();
    }
    const float mean = red[0] * (1.0f / SF_D);
    __syncthreads();

    const float d0 = x0 - mean, d1 = x1 - mean;
    red[tid] = d0 * d0 + d1 * d1;
    __syncthreads();
    for (int s = 128; s > 0; s >>= 1) {
        if (tid < s) red[tid] += red[tid + s];
        __syncthreads();
    }
    const float inv = rsqrtf(red[0] * (1.0f / SF_D) + SF_EPS);

    const float y0 = d0 * inv * g[tid] + be[tid];
    const float y1 = d1 * inv * g[tid + 256] + be[tid + 256];
    outf[base + tid] = y0;
    outf[base + tid + 256] = y1;
    outb[base + tid] = f2bf(y0);
    outb[base + tid + 256] = f2bf(y1);
}

// ---------------------------------------------------------------------------
// Transpose + convert: in f32 [R,C] -> out bf16 [C,R]
// ---------------------------------------------------------------------------
__global__ __launch_bounds__(256) void transpose_f32_bf16(
    const float* __restrict__ in, uint16_t* __restrict__ out, int R, int C)
{
    __shared__ float t[32][33];
    const int tx = threadIdx.x, ty = threadIdx.y;
    const int c  = blockIdx.x * 32 + tx;
    const int r0 = blockIdx.y * 32;
#pragma unroll
    for (int i = ty; i < 32; i += 8) {
        const int r = r0 + i;
        t[i][tx] = (r < R && c < C) ? in[(size_t)r * C + c] : 0.f;
    }
    __syncthreads();
    const int oc = r0 + tx;                // original row index
#pragma unroll
    for (int i = ty; i < 32; i += 8) {
        const int orr = blockIdx.x * 32 + i;   // original col index
        if (orr < C && oc < R)
            out[(size_t)orr * R + oc] = f2bf(t[tx][i]);
    }
}

__global__ __launch_bounds__(256) void f32_to_bf16_kernel(
    const float* __restrict__ in, uint16_t* __restrict__ out, size_t n)
{
    const size_t i = (size_t)blockIdx.x * blockDim.x + threadIdx.x;
    if (i < n) out[i] = f2bf(in[i]);
}

// ---------------------------------------------------------------------------
// Host orchestration
// ---------------------------------------------------------------------------
extern "C" void kernel_launch(void* const* d_in, const int* in_sizes, int n_in,
                              void* d_out, int out_size, void* d_ws, size_t ws_size,
                              hipStream_t stream)
{
    const float* x       = (const float*)d_in[0];
    const unsigned char* mask_r = (const unsigned char*)d_in[1];
    const unsigned char* mask_f = (const unsigned char*)d_in[2];
    const float* enc_w[5] = {(const float*)d_in[3], (const float*)d_in[4],
                             (const float*)d_in[5], (const float*)d_in[6],
                             (const float*)d_in[7]};
    const float* dec_w[5] = {(const float*)d_in[8], (const float*)d_in[9],
                             (const float*)d_in[10], (const float*)d_in[11],
                             (const float*)d_in[12]};
    const float* ff_w1 = (const float*)d_in[13];
    const float* ff_b1 = (const float*)d_in[14];
    const float* ff_w2 = (const float*)d_in[15];
    const float* ff_b2 = (const float*)d_in[16];
    const float* ln1_g = (const float*)d_in[17];
    const float* ln1_b = (const float*)d_in[18];
    const float* ln2_g = (const float*)d_in[19];
    const float* ln2_b = (const float*)d_in[20];
    const float* ln3_g = (const float*)d_in[21];
    const float* ln3_b = (const float*)d_in[22];
    const float* hw1   = (const float*)d_in[23];
    const float* hb1   = (const float*)d_in[24];
    const float* hw2   = (const float*)d_in[25];
    const float* hb2   = (const float*)d_in[26];

    const int N = SF_N, D = SF_D, FFN = SF_FFN, HID = SF_HID, OUT = SF_OUT;

    // workspace carve-out
    char* wsp = (char*)d_ws;
    size_t off = 0;
    auto alloc = [&](size_t bytes) -> void* {
        void* p = wsp + off;
        off = (off + bytes + 255) & ~(size_t)255;
        return p;
    };
    uint16_t* xb   = (uint16_t*)alloc((size_t)N * D * 2);
    uint16_t* Qr   = (uint16_t*)alloc((size_t)N * D * 2);
    uint16_t* Kr   = (uint16_t*)alloc((size_t)N * D * 2);
    uint16_t* Qf   = (uint16_t*)alloc((size_t)N * D * 2);
    uint16_t* Kf   = (uint16_t*)alloc((size_t)N * D * 2);
    uint16_t* Vt   = (uint16_t*)alloc((size_t)D * N * 2);
    uint16_t* wT[10];
    for (int i = 0; i < 10; ++i) wT[i] = (uint16_t*)alloc((size_t)D * D * 2);
    uint16_t* ffw1t = (uint16_t*)alloc((size_t)FFN * D * 2);
    uint16_t* ffw2t = (uint16_t*)alloc((size_t)D * FFN * 2);
    uint16_t* hw1t  = (uint16_t*)alloc((size_t)HID * D * 2);
    uint16_t* hw2t  = (uint16_t*)alloc((size_t)OUT * HID * 2);
    float*    Sr    = (float*)alloc((size_t)N * N * 4);
    float*    Sf    = (float*)alloc((size_t)N * N * 4);
    uint16_t* P     = (uint16_t*)alloc((size_t)N * N * 2);
    float*    AO    = (float*)alloc((size_t)N * D * 4);
    float*    h1    = (float*)alloc((size_t)N * D * 4);
    uint16_t* h1b   = (uint16_t*)alloc((size_t)N * D * 2);
    uint16_t* F1b   = (uint16_t*)alloc((size_t)N * FFN * 2);
    float*    F2    = (float*)alloc((size_t)N * D * 4);
    float*    h2    = (float*)alloc((size_t)N * D * 4);
    uint16_t* h2b   = (uint16_t*)alloc((size_t)N * D * 2);
    (void)ws_size; (void)n_in; (void)in_sizes; (void)out_size;

    const dim3 tb(32, 8);
    auto tgrid = [](int R, int C) { return dim3((C + 31) / 32, (R + 31) / 32); };

    // weight transposes (f32 [K,N] -> bf16 [N,K])
    for (int i = 0; i < 5; ++i)
        transpose_f32_bf16<<<tgrid(D, D), tb, 0, stream>>>(enc_w[i], wT[i], D, D);
    for (int i = 0; i < 5; ++i)
        transpose_f32_bf16<<<tgrid(D, D), tb, 0, stream>>>(dec_w[i], wT[5 + i], D, D);
    transpose_f32_bf16<<<tgrid(D, FFN), tb, 0, stream>>>(ff_w1, ffw1t, D, FFN);
    transpose_f32_bf16<<<tgrid(FFN, D), tb, 0, stream>>>(ff_w2, ffw2t, FFN, D);
    transpose_f32_bf16<<<tgrid(D, HID), tb, 0, stream>>>(hw1, hw1t, D, HID);
    transpose_f32_bf16<<<tgrid(HID, OUT), tb, 0, stream>>>(hw2, hw2t, HID, OUT);

    f32_to_bf16_kernel<<<((size_t)N * D + 255) / 256, 256, 0, stream>>>(x, xb, (size_t)N * D);

    auto ggrid = [](int M, int Nc) { return dim3((Nc + 127) / 128, M / 128); };

    // ---------------- dual attention block (shared for enc/dec) -------------
    auto dual_attn = [&](const uint16_t* inp, uint16_t* const* w5) {
        gemm_nt<EPI_BF16, 0><<<ggrid(N, D), 256, 0, stream>>>(inp, w5[0], Qr, nullptr, nullptr, 0.f, N, D, D, 0);
        gemm_nt<EPI_BF16, 0><<<ggrid(N, D), 256, 0, stream>>>(inp, w5[1], Kr, nullptr, nullptr, 0.f, N, D, D, 0);
        gemm_nt<EPI_BF16, 0><<<ggrid(N, D), 256, 0, stream>>>(inp, w5[2], Qf, nullptr, nullptr, 0.f, N, D, D, 0);
        gemm_nt<EPI_BF16, 0><<<ggrid(N, D), 256, 0, stream>>>(inp, w5[3], Kf, nullptr, nullptr, 0.f, N, D, D, 0);
        gemm_nt<EPI_TBF16, 0><<<ggrid(N, D), 256, 0, stream>>>(inp, w5[4], Vt, nullptr, nullptr, 0.f, N, D, D, N);
        gemm_nt<EPI_MASK, 0><<<ggrid(N, N), 256, 0, stream>>>(Qr, Kr, Sr, nullptr, mask_r, SF_SCALE, N, N, D, 0);
        gemm_nt<EPI_MASK, 0><<<ggrid(N, N), 256, 0, stream>>>(Qf, Kf, Sf, nullptr, mask_f, SF_SCALE, N, N, D, 0);
        softmax_dual<<<N, 256, 0, stream>>>(Sr, Sf, P);
        gemm_nt<EPI_F32, 0><<<ggrid(N, D), 256, 0, stream>>>(P, Vt, AO, nullptr, nullptr, 0.f, N, D, N, 0);
    };

    // encoder
    uint16_t* encT[5] = {wT[0], wT[1], wT[2], wT[3], wT[4]};
    dual_attn(xb, encT);
    ln_residual<<<N, 256, 0, stream>>>(AO, x, ln1_g, ln1_b, h1, h1b);

    // FFN
    gemm_nt<EPI_BF16, 1><<<ggrid(N, FFN), 256, 0, stream>>>(h1b, ffw1t, F1b, ff_b1, nullptr, 0.f, N, FFN, D, 0);
    gemm_nt<EPI_F32, 0><<<ggrid(N, D), 256, 0, stream>>>(F1b, ffw2t, F2, ff_b2, nullptr, 0.f, N, D, FFN, 0);
    ln_residual<<<N, 256, 0, stream>>>(h1, F2, ln2_g, ln2_b, h2, h2b);

    // decoder
    uint16_t* decT[5] = {wT[5], wT[6], wT[7], wT[8], wT[9]};
    dual_attn(h2b, decT);
    ln_residual<<<N, 256, 0, stream>>>(AO, h2, ln3_g, ln3_b, h1, h1b);  // recon -> h1/h1b

    // head: relu(recon @ hw1 + b1) @ hw2 + b2  -> d_out (f32, N x 64)
    gemm_nt<EPI_BF16, 2><<<ggrid(N, HID), 256, 0, stream>>>(h1b, hw1t, F1b, hb1, nullptr, 0.f, N, HID, D, 0);
    gemm_nt<EPI_F32, 0><<<ggrid(N, OUT), 256, 0, stream>>>(F1b, hw2t, (float*)d_out, hb2, nullptr, 0.f, N, OUT, HID, 0);
}
